// LexiconEncoder_68616397521541
// MI455X (gfx1250) — matrix operand order repaired and checked by reading, not compiled, not run on hardware
//
#include <hip/hip_runtime.h>

typedef __bf16 bf16;
typedef __attribute__((ext_vector_type(16))) __bf16   v16bf;
typedef __attribute__((ext_vector_type(8)))  float    v8f;
typedef __attribute__((ext_vector_type(4)))  unsigned u32x4;

union Frag16 { u32x4 q[2]; v16bf v; };

__device__ __forceinline__ bf16 f2bf(float f) {
  unsigned u = __builtin_bit_cast(unsigned, f);
  u += 0x7FFFu + ((u >> 16) & 1u);                 // round-to-nearest-even
  unsigned short h = (unsigned short)(u >> 16);
  return __builtin_bit_cast(bf16, h);
}

#define NB   64
#define ND   1024
#define NQ   128
#define NE   300
#define NEP  320      // K padded to multiple of 32
#define NH   256
#define NQT  304      // prealign N padded to multiple of 16
#define MDOC (NB * ND)            // 65536 doc rows
#define MALL (MDOC + NB * NQ)     // 73728 combined rows

// ---------------------------------------------------------------------------
// Prep: WT[n, k] = bf16(proj_w[k, n]), k padded 300..319 with zeros
__global__ void k_prep_w(const float* __restrict__ w, bf16* __restrict__ WT) {
  int idx = blockIdx.x * blockDim.x + threadIdx.x;   // 256*320 total
  if (idx >= NH * NEP) return;
  int n = idx / NEP, k = idx % NEP;
  float v = (k < NE) ? w[(size_t)k * NH + n] : 0.0f;
  WT[idx] = f2bf(v);
}

// ---------------------------------------------------------------------------
// Query embeddings: f32 out, bf16 X rows (after doc rows), bf16 QT (transposed)
__global__ void k_embed_query(const int* __restrict__ qtok,
                              const float* __restrict__ wemb,
                              float* __restrict__ out_q,
                              bf16* __restrict__ X,
                              bf16* __restrict__ QT) {
  int r = blockIdx.x;                 // 0 .. B*Q-1
  int b = r >> 7, q = r & 127;
  int tok = qtok[r];
  float s = (tok != 0) ? 1.0f : 0.0f;
  const float* src = wemb + (size_t)tok * NE;
  bf16* xrow = X + (size_t)(MDOC + r) * NEP;
  bf16* qt   = QT + (size_t)b * NQT * NQ + q;
  for (int e = threadIdx.x; e < NEP; e += blockDim.x) {
    float v = (e < NE) ? src[e] * s : 0.0f;
    if (e < NE) out_q[(size_t)r * NE + e] = v;
    bf16 h = f2bf(v);
    xrow[e] = h;
    if (e < NQT) qt[(size_t)e * NQ] = h;
  }
}

// ---------------------------------------------------------------------------
// Doc embeddings: f32 into out cols [0,300), bf16 X rows; ner/pos/fea cols 600..623
__global__ void k_embed_doc(const int* __restrict__ dtok,
                            const int* __restrict__ dpos,
                            const int* __restrict__ dner,
                            const float* __restrict__ dfea,
                            const float* __restrict__ wemb,
                            const float* __restrict__ pemb,
                            const float* __restrict__ nemb,
                            float* __restrict__ out_doc,
                            bf16* __restrict__ X) {
  int r = blockIdx.x;                 // 0 .. B*D-1
  int tok = dtok[r];
  float s = (tok != 0) ? 1.0f : 0.0f;
  const float* src = wemb + (size_t)tok * NE;
  float* orow = out_doc + (size_t)r * 624;
  bf16* xrow  = X + (size_t)r * NEP;
  for (int e = threadIdx.x; e < NEP; e += blockDim.x) {
    float v = (e < NE) ? src[e] * s : 0.0f;
    if (e < NE) orow[e] = v;
    xrow[e] = f2bf(v);
  }
  int t = threadIdx.x;
  if (t < 8) {
    int id = dner[r];
    orow[600 + t] = nemb[id * 8 + t] * ((id != 0) ? 1.0f : 0.0f);
  } else if (t < 20) {
    int id = dpos[r], j = t - 8;
    orow[608 + j] = pemb[id * 12 + j] * ((id != 0) ? 1.0f : 0.0f);
  } else if (t < 24) {
    int j = t - 20;
    orow[620 + j] = dfea[(size_t)r * 4 + j];
  }
}

// ---------------------------------------------------------------------------
// GEMM1: P = relu(X @ W + b) in bf16.  M=73728, N=256, K=320.
// Block = 4 waves; wave w -> cols [64w, 64w+64), strip = 16 rows.
// Weight K-slices are staged in LDS via GLOBAL_LOAD_ASYNC_TO_LDS_B128
// (ASYNCcnt path), then read as ds_load_b128 fragments.
__global__ void __launch_bounds__(128) k_proj(const bf16* __restrict__ X,
                                              const bf16* __restrict__ WT,
                                              const float* __restrict__ bias,
                                              bf16* __restrict__ P) {
  __shared__ bf16 ldsB[NH * 32];     // 256 rows x 32 k  = 16 KB per slice
  int lane = threadIdx.x & 31, wave = threadIdx.x >> 5;
  int m16 = lane & 15, hf = lane >> 4;
  int row0 = blockIdx.x * 16;
  int nbase = wave * 64;
  const bf16* arow = X + (size_t)(row0 + m16) * NEP;
  unsigned lds_base = (unsigned)(uintptr_t)(&ldsB[0]);   // LDS offset = addr[31:0]

  v8f acc[4];
#pragma unroll
  for (int t = 0; t < 4; ++t)
#pragma unroll
    for (int r = 0; r < 8; ++r) acc[t][r] = 0.0f;

  for (int kc = 0; kc < NEP; kc += 32) {
    // ---- async stage WT[:, kc:kc+32] -> LDS (256 x 64B, 16B per lane) ----
#pragma unroll
    for (int it = 0; it < 8; ++it) {
      int idx  = it * 128 + threadIdx.x;     // 0..1023 16B-chunks
      int n    = idx >> 2;
      int part = idx & 3;
      unsigned loff = lds_base + (unsigned)(n * 64 + part * 16);
      unsigned long long gaddr =
          (unsigned long long)(uintptr_t)(WT + (size_t)n * NEP + kc + part * 8);
      asm volatile("global_load_async_to_lds_b128 %0, %1, off"
                   :: "v"(loff), "v"(gaddr) : "memory");
    }
    asm volatile("s_wait_asynccnt 0x0" ::: "memory");
    __syncthreads();

    __builtin_prefetch(arow + kc + 64, 0, 0);
    Frag16 a;
    a.q[0] = *(const u32x4*)(arow + kc + hf * 16);
    a.q[1] = *(const u32x4*)(arow + kc + hf * 16 + 8);
#pragma unroll
    for (int t = 0; t < 4; ++t) {
      const bf16* brow = &ldsB[(nbase + t * 16 + m16) * 32 + hf * 16];
      Frag16 bb;
      bb.q[0] = *(const u32x4*)(brow);       // ds_load_b128
      bb.q[1] = *(const u32x4*)(brow + 8);   // ds_load_b128
      acc[t] = __builtin_amdgcn_wmma_f32_16x16x32_bf16(
          false, a.v, false, bb.v, (short)0, acc[t], false, false);
    }
    __syncthreads();                         // before next slice overwrites LDS
  }
#pragma unroll
  for (int t = 0; t < 4; ++t) {
    int n = nbase + t * 16 + m16;
    float bv = bias[n];
#pragma unroll
    for (int r = 0; r < 8; ++r) {
      float v = fmaxf(acc[t][r] + bv, 0.0f);
      P[(size_t)(row0 + r + 8 * hf) * NH + n] = f2bf(v);
    }
  }
}

// ---------------------------------------------------------------------------
// GEMM2 + fused softmax: per batch S = Pd @ Pq^T (1024x256x128), softmax over q,
// write attn bf16.  Wave owns full 16x128 rows -> in-register softmax.
__global__ void __launch_bounds__(128) k_scores(const bf16* __restrict__ P,
                                                bf16* __restrict__ ATT) {
  int lane = threadIdx.x & 31, wave = threadIdx.x >> 5;
  int m16 = lane & 15, hf = lane >> 4;
  int gs = blockIdx.x * 4 + wave;      // 0..4095
  int b = gs >> 6, s = gs & 63;
  int rowA0 = b * ND + s * 16;
  int rowB0 = MDOC + b * NQ;
  const bf16* arow = P + (size_t)(rowA0 + m16) * NH;

  v8f acc[8];
#pragma unroll
  for (int t = 0; t < 8; ++t)
#pragma unroll
    for (int r = 0; r < 8; ++r) acc[t][r] = 0.0f;

  for (int kc = 0; kc < NH; kc += 32) {
    Frag16 a;
    a.q[0] = *(const u32x4*)(arow + kc + hf * 16);
    a.q[1] = *(const u32x4*)(arow + kc + hf * 16 + 8);
#pragma unroll
    for (int t = 0; t < 8; ++t) {
      const bf16* brow = P + (size_t)(rowB0 + t * 16 + m16) * NH + kc + hf * 16;
      Frag16 bb;
      bb.q[0] = *(const u32x4*)(brow);
      bb.q[1] = *(const u32x4*)(brow + 8);
      acc[t] = __builtin_amdgcn_wmma_f32_16x16x32_bf16(
          false, a.v, false, bb.v, (short)0, acc[t], false, false);
    }
  }
  // softmax over the 128 q-values of each row (8 tiles x 16 lanes per half)
#pragma unroll
  for (int r = 0; r < 8; ++r) {
    float mx = -3.402823466e38f;
#pragma unroll
    for (int t = 0; t < 8; ++t) mx = fmaxf(mx, acc[t][r]);
    for (int off = 1; off < 16; off <<= 1)
      mx = fmaxf(mx, __shfl_xor(mx, off, 32));   // masks <16 stay in half-group
    float sum = 0.0f;
#pragma unroll
    for (int t = 0; t < 8; ++t) {
      float e = __expf(acc[t][r] - mx);
      acc[t][r] = e;
      sum += e;
    }
    for (int off = 1; off < 16; off <<= 1)
      sum += __shfl_xor(sum, off, 32);
    float inv = 1.0f / sum;
    int row = rowA0 + r + 8 * hf;
#pragma unroll
    for (int t = 0; t < 8; ++t)
      ATT[(size_t)row * NQ + t * 16 + m16] = f2bf(acc[t][r] * inv);
  }
}

// ---------------------------------------------------------------------------
// GEMM3: prealign = attn @ query_e (per batch 1024x128x300), f32 stores into
// out cols [300,600).  A-fragments cached across all 19 N-tiles.
__global__ void __launch_bounds__(128) k_prealign(const bf16* __restrict__ ATT,
                                                  const bf16* __restrict__ QT,
                                                  float* __restrict__ out_doc) {
  int lane = threadIdx.x & 31, wave = threadIdx.x >> 5;
  int m16 = lane & 15, hf = lane >> 4;
  int gs = blockIdx.x * 4 + wave;
  int b = gs >> 6, s = gs & 63;
  int rowA0 = b * ND + s * 16;
  const bf16* arow = ATT + (size_t)(rowA0 + m16) * NQ;
  const bf16* qtb  = QT + (size_t)b * NQT * NQ;

  u32x4 aq[4][2];
#pragma unroll
  for (int c = 0; c < 4; ++c) {
    aq[c][0] = *(const u32x4*)(arow + c * 32 + hf * 16);
    aq[c][1] = *(const u32x4*)(arow + c * 32 + hf * 16 + 8);
  }
  for (int t = 0; t < 19; ++t) {       // uniform loop; EXEC stays all-ones
    v8f acc;
#pragma unroll
    for (int r = 0; r < 8; ++r) acc[r] = 0.0f;
#pragma unroll
    for (int c = 0; c < 4; ++c) {
      Frag16 a;
      a.q[0] = aq[c][0];
      a.q[1] = aq[c][1];
      const bf16* brow = qtb + (size_t)(t * 16 + m16) * NQ + c * 32 + hf * 16;
      Frag16 bb;
      bb.q[0] = *(const u32x4*)(brow);
      bb.q[1] = *(const u32x4*)(brow + 8);
      acc = __builtin_amdgcn_wmma_f32_16x16x32_bf16(
          false, a.v, false, bb.v, (short)0, acc, false, false);
    }
    int n = t * 16 + m16;
    if (n < NE) {
#pragma unroll
      for (int r = 0; r < 8; ++r)
        out_doc[(size_t)(rowA0 + r + 8 * hf) * 624 + 300 + n] = acc[r];
    }
  }
}

// ---------------------------------------------------------------------------
extern "C" void kernel_launch(void* const* d_in, const int* in_sizes, int n_in,
                              void* d_out, int out_size, void* d_ws, size_t ws_size,
                              hipStream_t stream) {
  const int*   qtok = (const int*)d_in[0];
  const int*   dtok = (const int*)d_in[1];
  const int*   dpos = (const int*)d_in[2];
  const int*   dner = (const int*)d_in[3];
  const float* dfea = (const float*)d_in[4];
  const float* wemb = (const float*)d_in[5];
  const float* pw   = (const float*)d_in[6];
  const float* pb   = (const float*)d_in[7];
  const float* pemb = (const float*)d_in[8];
  const float* nemb = (const float*)d_in[9];

  float* out_q   = (float*)d_out;                       // [B,Q,300]
  float* out_doc = out_q + (size_t)NB * NQ * NE;        // [B,D,624]

  char* ws = (char*)d_ws;
  bf16* X   = (bf16*)(ws + 0);            // 73728*320*2 = 47,185,920 B
  bf16* P   = (bf16*)(ws + 47185920);     // 73728*256*2 = 37,748,736 B
  bf16* ATT = (bf16*)(ws + 84934656);     // 65536*128*2 = 16,777,216 B
  bf16* WT  = (bf16*)(ws + 101711872);    // 256*320*2   =    163,840 B
  bf16* QT  = (bf16*)(ws + 101875712);    // 64*304*128*2 = 4,980,736 B

  k_prep_w     <<<NEP, 256, 0, stream>>>(pw, WT);
  k_embed_query<<<NB * NQ, 128, 0, stream>>>(qtok, wemb, out_q, X, QT);
  k_embed_doc  <<<NB * ND, 128, 0, stream>>>(dtok, dpos, dner, dfea, wemb,
                                             pemb, nemb, out_doc, X);
  k_proj       <<<MALL / 16, 128, 0, stream>>>(X, WT, pb, P);
  k_scores     <<<(NB * ND / 16) / 4, 128, 0, stream>>>(P, ATT);
  k_prealign   <<<(NB * ND / 16) / 4, 128, 0, stream>>>(ATT, QT, out_doc);
}
